// Attention_68771016343741
// MI455X (gfx1250) — matrix operand order repaired and checked by reading, not compiled, not run on hardware
//
#include <hip/hip_runtime.h>
#include <stdint.h>
#include <stddef.h>

// Problem constants (match reference)
#define BQ   4
#define SLQ  1024
#define SLK  1024
#define DIM  1024
#define NH   16
#define DHD  64

typedef __bf16 v16bf __attribute__((ext_vector_type(16)));
typedef float  v8f   __attribute__((ext_vector_type(8)));

union Frag16 { v16bf v; uint32_t u[8]; };

__device__ __forceinline__ uint16_t f32_to_bf16(float f) {
    uint32_t u = __float_as_uint(f);
    uint32_t rnd = 0x7FFFu + ((u >> 16) & 1u);
    return (uint16_t)((u + rnd) >> 16);
}

// A-operand (16xK) 16-bit: dword index for VGPR j / half-wave
__device__ __forceinline__ int dwA(int j, int half) {
    return ((j < 4) ? j : (8 + (j - 4))) + 4 * half;
}
// B-operand (Kx16) 16-bit: dword index for VGPR j / half-wave
__device__ __forceinline__ int dwB(int j, int half) {
    return ((j < 4) ? j : (8 + (j - 4))) + 4 * half;
}

// CDNA5 async global->LDS copy (16B per lane), tracked by ASYNCcnt.
__device__ __forceinline__ void async_ld_b128(uint32_t lds_off, const void* gaddr) {
    uint64_t ga = (uint64_t)(uintptr_t)gaddr;
    asm volatile("global_load_async_to_lds_b128 %0, %1, off"
                 :: "v"(lds_off), "v"(ga) : "memory");
}
__device__ __forceinline__ uint32_t lds_off_of(const void* p) {
    return (uint32_t)(uintptr_t)p;   // LDS aperture: addr[31:0] == LDS offset
}
#define WAIT_ASYNC(n) asm volatile("s_wait_asynccnt %0" :: "i"(n) : "memory")

// ---------------------------------------------------------------------------
// fp32 -> bf16 convert (activations)
__global__ __launch_bounds__(256) void cvt_bf16(const float* __restrict__ in,
                                                uint16_t* __restrict__ out, int n) {
    int i = blockIdx.x * 256 + threadIdx.x;
    if (i < n) out[i] = f32_to_bf16(in[i]);
}

// fp32 [K,N] -> bf16 transposed [N,K] (weights)
__global__ __launch_bounds__(256) void cvt_w_tr(const float* __restrict__ W,
                                                uint16_t* __restrict__ Wt,
                                                int Kd, int Nd) {
    int i = blockIdx.x * 256 + threadIdx.x;
    if (i < Kd * Nd) {
        int k = i / Nd, n = i % Nd;
        Wt[(size_t)n * Kd + k] = f32_to_bf16(W[i]);
    }
}

// ---------------------------------------------------------------------------
// bf16 WMMA GEMM: C[M,N] = A[M,K] @ Bt[N,K]^T + bias, fp32 accumulate.
// B tile (64 cols x 64 k = 8KB) shared by all 8 waves via double-buffered
// async LDS staging. 8 WMMAs per wave between barriers; all fragment loads
// batched ahead of the WMMA group.
// MODE 0: bf16 -> [B,H,L,DH] (Q/K projection), scaled
// MODE 1: fp32 row-major [M,N] (final output projection)
// MODE 2: bf16 -> [B,H,DH,LK] (V projection, per-head transposed)
#define BROW 36   // dwords per staged B row (128B data + 16B pad, conflict-free)
template <int MODE>
__global__ __launch_bounds__(256) void gemm_bf16(const uint16_t* __restrict__ A,
                                                 const uint16_t* __restrict__ Bt,
                                                 const float* __restrict__ bias,
                                                 void* __restrict__ Cout,
                                                 int M, int N, int K, float scale) {
    __shared__ __align__(16) uint32_t bbuf[2][64 * BROW];  // 2 x 9 KiB

    const int tid  = threadIdx.x;
    const int lane = tid & 31;
    const int wave = tid >> 5;             // 0..7
    const int n16  = lane & 15;
    const int half = lane >> 4;

    const int rowBase = blockIdx.y * 128 + wave * 16;
    const int colBase = blockIdx.x * 64;
    const int arow_m  = rowBase + n16;

    // staging: 512 x 16B segments (64 rows x 128B), 2 per thread
    auto stage = [&](int buf, int k0) {
#pragma unroll
        for (int s = 0; s < 2; ++s) {
            const int seg = tid + s * 256;           // 0..511
            const int row = seg >> 3, ss = seg & 7;  // row 0..63, 8 segs/row
            async_ld_b128(lds_off_of(&bbuf[buf][row * BROW + ss * 4]),
                          Bt + (size_t)(colBase + row) * K + k0 + ss * 8);
        }
    };

    v8f acc[4] = {};

    const int nsteps = K / 64;
    stage(0, 0);                                     // prologue

    for (int step = 0; step < nsteps; ++step) {
        const int k0 = step * 64;
        const int cur = step & 1;
        if (step + 1 < nsteps) {
            stage(cur ^ 1, k0 + 64);
            WAIT_ASYNC(2);                           // current tile complete
        } else {
            WAIT_ASYNC(0);
        }
        __syncthreads();

        // A fragments for both 32-deep K chunks (one global load clause)
        Frag16 a[2];
        const uint32_t* arow = (const uint32_t*)(A + (size_t)arow_m * K + k0);
#pragma unroll
        for (int c = 0; c < 2; ++c)
#pragma unroll
            for (int j = 0; j < 8; ++j) a[c].u[j] = arow[c * 16 + dwA(j, half)];

        // All 8 B fragments first (one DS load clause), then 8 WMMAs
        Frag16 bf[2][4];
#pragma unroll
        for (int t = 0; t < 4; ++t) {
            const uint32_t* brow = &bbuf[cur][(t * 16 + n16) * BROW];
#pragma unroll
            for (int c = 0; c < 2; ++c)
#pragma unroll
                for (int j = 0; j < 8; ++j)
                    bf[c][t].u[j] = brow[c * 16 + dwB(j, half)];
        }
#pragma unroll
        for (int c = 0; c < 2; ++c)
#pragma unroll
            for (int t = 0; t < 4; ++t)
                acc[t] = __builtin_amdgcn_wmma_f32_16x16x32_bf16(
                    false, a[c].v, false, bf[c][t].v, (short)0, acc[t], false, false);

        __syncthreads();                             // done reading cur buffer
    }

    // Epilogue: C layout lane n = lane%16, row m = j + 8*half
#pragma unroll
    for (int t = 0; t < 4; ++t) {
        const int c = colBase + t * 16 + n16;
        const float bv = bias[c];
#pragma unroll
        for (int j = 0; j < 8; ++j) {
            const int r = rowBase + j + 8 * half;
            const float v = (acc[t][j] + bv) * scale;
            if (MODE == 1) {
                ((float*)Cout)[(size_t)r * N + c] = v;
            } else {
                const int bb = r / SLQ, l = r % SLQ;
                const int h = c / DHD, d = c % DHD;
                uint16_t* o = (uint16_t*)Cout;
                if (MODE == 0)
                    o[(((size_t)bb * NH + h) * SLQ + l) * DHD + d] = f32_to_bf16(v);
                else // MODE 2
                    o[(((size_t)bb * NH + h) * DHD + d) * SLK + l] = f32_to_bf16(v);
            }
        }
    }
}

// ---------------------------------------------------------------------------
// Flash attention. Block = 4 waves = 4 q-tiles of the same (b,h).
// K tile (32x64) and V tile (64x32) async-staged into double-buffered LDS,
// shared by the 4 waves. All fragment loads batched ahead of WMMA groups.
#define KPAD 36   // dwords per staged K row (128B data + 16B pad)
#define VPAD 20   // dwords per staged V row (64B data + 16B pad)
#define PPAD 40   // uint16 per P row (64B data + 16B pad)
__global__ __launch_bounds__(128) void attn_kernel(const uint16_t* __restrict__ Qs,
                                                   const uint16_t* __restrict__ Ks,
                                                   const uint16_t* __restrict__ Vt,
                                                   const int* __restrict__ qmask,
                                                   const int* __restrict__ kmask,
                                                   uint16_t* __restrict__ Oa) {
    __shared__ __align__(16) uint32_t kbuf[2][32 * KPAD];   // 2 x 4.5 KiB
    __shared__ __align__(16) uint32_t vbuf[2][64 * VPAD];   // 2 x 5 KiB
    __shared__ __align__(16) uint16_t pbuf[4][16 * PPAD];   // 4 x 1.25 KiB

    const int tid  = threadIdx.x;
    const int lane = tid & 31;
    const int wib  = tid >> 5;                  // wave in block 0..3
    const int gw   = blockIdx.x * 4 + wib;      // global wave id
    const int qt   = gw & 63;                   // q tile (block spans 4 of same h)
    const int h    = (gw >> 6) & (NH - 1);
    const int b    = gw >> 10;
    const int n16  = lane & 15;
    const int half = lane >> 4;

    const uint16_t* qbase = Qs + (((size_t)b * NH + h) * SLQ + qt * 16) * DHD;
    const uint16_t* kb_p  = Ks + ((size_t)b * NH + h) * SLK * DHD;
    const uint16_t* vbase = Vt + ((size_t)b * NH + h) * DHD * SLK;

    auto stage = [&](int buf, int key0) {
#pragma unroll
        for (int s = 0; s < 2; ++s) {
            const int kseg = tid + s * 128;     // 0..255
            const int krow = kseg >> 3, kss = kseg & 7;
            async_ld_b128(lds_off_of(&kbuf[buf][krow * KPAD + kss * 4]),
                          kb_p + (size_t)(key0 + krow) * DHD + kss * 8);
        }
#pragma unroll
        for (int s = 0; s < 2; ++s) {
            const int vseg = tid + s * 128;
            const int vrow = vseg >> 2, vss = vseg & 3;
            async_ld_b128(lds_off_of(&vbuf[buf][vrow * VPAD + vss * 4]),
                          vbase + (size_t)vrow * SLK + key0 + vss * 8);
        }
    };

    // Q A-fragments: two K-chunks covering DH=64 (global; reused 32 iters)
    Frag16 qf[2];
    {
        const uint32_t* qrow = (const uint32_t*)(qbase + (size_t)n16 * DHD);
#pragma unroll
        for (int c = 0; c < 2; ++c)
#pragma unroll
            for (int j = 0; j < 8; ++j) qf[c].u[j] = qrow[c * 16 + dwA(j, half)];
    }

    float rmax[8], rsum[8];
    v8f o[4] = {};
#pragma unroll
    for (int j = 0; j < 8; ++j) { rmax[j] = -1e30f; rsum[j] = 0.0f; }

    const int nsteps = SLK / 32;
    stage(0, 0);                                 // prologue

    for (int it = 0; it < nsteps; ++it) {
        const int key0 = it * 32;
        const int cur = it & 1;
        if (it + 1 < nsteps) {
            stage(cur ^ 1, key0 + 32);
            WAIT_ASYNC(4);                       // current tile's 4 copies done
        } else {
            WAIT_ASYNC(0);
        }
        // hoist key-mask loads so they overlap the score WMMAs
        const int km0 = kmask[b * SLK + key0 + n16];
        const int km1 = kmask[b * SLK + key0 + 16 + n16];
        __syncthreads();

        // ---- all 4 K fragments first, then 4 score WMMAs
        Frag16 kf[2][2];                         // [st][chunk]
#pragma unroll
        for (int st = 0; st < 2; ++st) {
            const uint32_t* krow = &kbuf[cur][(st * 16 + n16) * KPAD];
#pragma unroll
            for (int c = 0; c < 2; ++c)
#pragma unroll
                for (int j = 0; j < 8; ++j)
                    kf[st][c].u[j] = krow[c * 16 + dwB(j, half)];
        }
        v8f s0 = {}, s1 = {};
#pragma unroll
        for (int c = 0; c < 2; ++c) {
            s0 = __builtin_amdgcn_wmma_f32_16x16x32_bf16(
                false, qf[c].v, false, kf[0][c].v, (short)0, s0, false, false);
            s1 = __builtin_amdgcn_wmma_f32_16x16x32_bf16(
                false, qf[c].v, false, kf[1][c].v, (short)0, s1, false, false);
        }

        // ---- additive key mask (column = key index held by this lane)
#pragma unroll
        for (int j = 0; j < 8; ++j) {
            if (!km0) s0[j] = -1e30f;
            if (!km1) s1[j] = -1e30f;
        }

        // ---- online softmax (row m = j + 8*half, spread over 16 lanes)
#pragma unroll
        for (int j = 0; j < 8; ++j) {
            float mloc = fmaxf(s0[j], s1[j]);
#pragma unroll
            for (int m = 8; m >= 1; m >>= 1)
                mloc = fmaxf(mloc, __shfl_xor(mloc, m, 32));
            const float nmax = fmaxf(rmax[j], mloc);
            const float corr = __expf(rmax[j] - nmax);
            const float p0 = __expf(s0[j] - nmax);
            const float p1 = __expf(s1[j] - nmax);
            float sl = p0 + p1;
#pragma unroll
            for (int m = 8; m >= 1; m >>= 1) sl += __shfl_xor(sl, m, 32);
            rsum[j] = rsum[j] * corr + sl;
            rmax[j] = nmax;
#pragma unroll
            for (int t = 0; t < 4; ++t) o[t][j] *= corr;

            const int r = j + 8 * half;
            pbuf[wib][r * PPAD + n16]      = f32_to_bf16(p0);
            pbuf[wib][r * PPAD + 16 + n16] = f32_to_bf16(p1);
        }
        asm volatile("s_wait_dscnt 0" ::: "memory");

        // ---- reload P in A-fragment layout (16x32)
        Frag16 pf;
        {
            const uint32_t* prow = (const uint32_t*)&pbuf[wib][n16 * PPAD];
#pragma unroll
            for (int j = 0; j < 8; ++j) pf.u[j] = prow[dwA(j, half)];
        }

        // ---- all 4 V fragments first, then 4 PV WMMAs
        Frag16 vf[4];
#pragma unroll
        for (int t = 0; t < 4; ++t) {
            const uint32_t* vrow = &vbuf[cur][(t * 16 + n16) * VPAD];
#pragma unroll
            for (int j = 0; j < 8; ++j) vf[t].u[j] = vrow[dwB(j, half)];
        }
#pragma unroll
        for (int t = 0; t < 4; ++t)
            o[t] = __builtin_amdgcn_wmma_f32_16x16x32_bf16(
                false, pf.v, false, vf[t].v, (short)0, o[t], false, false);

        __syncthreads();                          // all waves done with cur tile
    }

    // ---- finalize: divide by rowsum, multiplicative query mask, store bf16
#pragma unroll
    for (int j = 0; j < 8; ++j) {
        const int qrow = qt * 16 + j + 8 * half;
        const float inv = 1.0f / rsum[j];
        const float qm = (float)qmask[b * SLQ + qrow];
#pragma unroll
        for (int t = 0; t < 4; ++t) {
            const float v = o[t][j] * inv * qm;
            Oa[((size_t)b * SLQ + qrow) * (NH * DHD) + h * DHD + t * 16 + n16] =
                f32_to_bf16(v);
        }
    }
}

// ---------------------------------------------------------------------------
extern "C" void kernel_launch(void* const* d_in, const int* in_sizes, int n_in,
                              void* d_out, int out_size, void* d_ws, size_t ws_size,
                              hipStream_t stream) {
    (void)in_sizes; (void)n_in; (void)out_size; (void)ws_size;

    const float* q      = (const float*)d_in[0];
    const float* x      = (const float*)d_in[1];
    const int*   q_mask = (const int*)d_in[2];
    const int*   k_mask = (const int*)d_in[3];
    const float* Wq     = (const float*)d_in[4];
    const float* bq     = (const float*)d_in[5];
    const float* Wk     = (const float*)d_in[6];
    const float* bk     = (const float*)d_in[7];
    const float* Wv     = (const float*)d_in[8];
    const float* bv     = (const float*)d_in[9];
    const float* Wo     = (const float*)d_in[10];
    const float* bo     = (const float*)d_in[11];

    char* ws = (char*)d_ws;
    size_t off = 0;
    const size_t actB = (size_t)BQ * SLQ * DIM * sizeof(uint16_t);  // 8 MiB
    const size_t wB   = (size_t)DIM * DIM * sizeof(uint16_t);       // 2 MiB
    uint16_t* qbf = (uint16_t*)(ws + off); off += actB;
    uint16_t* xbf = (uint16_t*)(ws + off); off += actB;
    uint16_t* Wqt = (uint16_t*)(ws + off); off += wB;
    uint16_t* Wkt = (uint16_t*)(ws + off); off += wB;
    uint16_t* Wvt = (uint16_t*)(ws + off); off += wB;
    uint16_t* Wot = (uint16_t*)(ws + off); off += wB;
    uint16_t* Qs  = (uint16_t*)(ws + off); off += actB;
    uint16_t* Ks  = (uint16_t*)(ws + off); off += actB;
    uint16_t* Vt  = (uint16_t*)(ws + off); off += actB;
    uint16_t* Oa  = (uint16_t*)(ws + off); off += actB;

    const int nact = BQ * SLQ * DIM;
    cvt_bf16<<<(nact + 255) / 256, 256, 0, stream>>>(q, qbf, nact);
    cvt_bf16<<<(nact + 255) / 256, 256, 0, stream>>>(x, xbf, nact);
    const int nw = DIM * DIM;
    cvt_w_tr<<<(nw + 255) / 256, 256, 0, stream>>>(Wq, Wqt, DIM, DIM);
    cvt_w_tr<<<(nw + 255) / 256, 256, 0, stream>>>(Wk, Wkt, DIM, DIM);
    cvt_w_tr<<<(nw + 255) / 256, 256, 0, stream>>>(Wv, Wvt, DIM, DIM);
    cvt_w_tr<<<(nw + 255) / 256, 256, 0, stream>>>(Wo, Wot, DIM, DIM);

    const int M = BQ * SLQ;                  // 4096
    dim3 gg(DIM / 64, M / 128);              // 16 x 32 blocks
    // Q projection: fold 1/sqrt(DH)=0.125 into stored Q
    gemm_bf16<0><<<gg, 256, 0, stream>>>(qbf, Wqt, bq, Qs, M, DIM, DIM, 0.125f);
    gemm_bf16<0><<<gg, 256, 0, stream>>>(xbf, Wkt, bk, Ks, M, DIM, DIM, 1.0f);
    gemm_bf16<2><<<gg, 256, 0, stream>>>(xbf, Wvt, bv, Vt, M, DIM, DIM, 1.0f);

    const int nwaves = BQ * NH * (SLQ / 16); // 4096 waves
    attn_kernel<<<nwaves / 4, 128, 0, stream>>>(Qs, Ks, Vt, q_mask, k_mask, Oa);

    gemm_bf16<1><<<gg, 256, 0, stream>>>(Oa, Wot, bo, (float*)d_out, M, DIM, DIM, 1.0f);
}
